// AnomalyAttention_51402168598811
// MI455X (gfx1250) — matrix operand order repaired and checked by reading, not compiled, not run on hardware
//
#include <hip/hip_runtime.h>
#include <math.h>

typedef _Float16 v16h __attribute__((ext_vector_type(16)));
typedef _Float16 h8   __attribute__((ext_vector_type(8)));
typedef _Float16 h2   __attribute__((ext_vector_type(2)));
typedef float    v8f  __attribute__((ext_vector_type(8)));

#define LN3 1.0986122886681098f
#define LOG2E 1.4426950408889634f
#define INV_SQRT_2PI 0.3989422804014327f
#define NEG_BIG (-3.0e38f)
// scale (1/sqrt(64)) * log2(e), folded into the Q fragment so WMMA emits
// base-2 logits directly.
#define QSCALE (0.125f * LOG2E)

// Raw hardware transcendentals: v_exp_f32 is natively 2^x, v_log_f32 is log2.
__device__ __forceinline__ float fexp2(float x) {
  return __builtin_amdgcn_exp2f(x);
}
__device__ __forceinline__ float flog2(float x) {
  return __builtin_amdgcn_logf(x);
}

// DPP16 lane moves (v_mov_b32_dpp): no LDS traffic, no dscnt waits.
template <int CTRL>
__device__ __forceinline__ float dppmov(float v) {
  return __int_as_float(__builtin_amdgcn_update_dpp(
      0, __float_as_int(v), CTRL, 0xF, 0xF, true));
}
// Reduce across each 16-lane half-row (lanes 0-15 and 16-31 independently):
// quad xor-1, quad xor-2, then row_ror:4 and row_ror:8 combine the 4 quads.
__device__ __forceinline__ float rowmax16(float v) {
  v = fmaxf(v, dppmov<0xB1>(v));   // quad_perm(1,0,3,2)
  v = fmaxf(v, dppmov<0x4E>(v));   // quad_perm(2,3,0,1)
  v = fmaxf(v, dppmov<0x124>(v));  // row_ror:4
  v = fmaxf(v, dppmov<0x128>(v));  // row_ror:8
  return v;
}
__device__ __forceinline__ float rowsum16(float v) {
  v += dppmov<0xB1>(v);
  v += dppmov<0x4E>(v);
  v += dppmov<0x124>(v);
  v += dppmov<0x128>(v);
  return v;
}

// ---------------------------------------------------------------------------
// Kernel 1: sg[b,h,l] = 3^(sigmoid(5*sigma)+1e-5) - 1   (sigma is [B,L,H])
// ---------------------------------------------------------------------------
__global__ void sg_precompute_kernel(const float* __restrict__ sigma,
                                     float* __restrict__ sg,
                                     int B, int L, int H) {
  int idx = blockIdx.x * blockDim.x + threadIdx.x;
  int total = B * H * L;
  if (idx >= total) return;
  int l  = idx % L;
  int bh = idx / L;
  int h  = bh % H;
  int b  = bh / H;
  float x = sigma[((size_t)b * L + l) * H + h];
  float s = 1.0f / (1.0f + fexp2(-5.0f * LOG2E * x)) + 1e-5f;  // sigmoid(5x)
  sg[idx] = fexp2(s * (LN3 * LOG2E)) - 1.0f;                   // 3^s - 1
}

// ---------------------------------------------------------------------------
// Kernel 2: flash-style causal attention with multiplicative Gaussian kernel.
// Block = 4 wave32 waves; each wave owns 16 query rows (64 rows / block).
// All softmax math is done in exp2 domain (v_exp_f32 is natively 2^x).
// ---------------------------------------------------------------------------
__global__ __launch_bounds__(128) void anomaly_attn_kernel(
    const float* __restrict__ Q, const float* __restrict__ Km,
    const float* __restrict__ Vm, const float* __restrict__ sg_arr,
    float* __restrict__ Out, int B, int L, int H) {
  constexpr int E = 64;
  constexpr int KT_STRIDE = 80;  // f16/row: 160B rows -> 16B-aligned b128 reads
  constexpr int VT_STRIDE = 40;  // f16/row: 80B rows  -> 16B-aligned
  constexpr int PW_STRIDE = 40;  // f16/row: 80B rows  -> 16B-aligned

  __shared__ __align__(16) _Float16 Kt[32 * KT_STRIDE];     // K tile  [key][e]
  __shared__ __align__(16) _Float16 VtT[64 * VT_STRIDE];    // V tileT [e][key]
  __shared__ __align__(16) _Float16 Pw[4][16 * PW_STRIDE];  // per-wave P stage

  const int bh = blockIdx.y;
  const int b = bh / H, h = bh % H;
  const int tileT = blockIdx.x;        // 64-row query tile
  const int wave = threadIdx.x >> 5;
  const int lane = threadIdx.x & 31;
  const int lane16 = lane & 15;
  const int hi = lane >> 4;            // 0: lanes 0-15, 1: lanes 16-31
  const int qbase = tileT * 64 + wave * 16;

  // A-fragment element->K mapping (contiguous K = 16*hi + i, matched by the B reads)
  const int off1 = hi ? 16 : 0;        // elems [0..7]  -> K = off1+i
  const int off2 = hi ? 24 : 8;        // elems [8..15] -> K = off2+(i-8)

  // ---- staging coordinates: each thread owns a row-pair x 8-wide e chunk ---
  const int rp = threadIdx.x >> 3;       // 0..15 (key row pair 2rp,2rp+1)
  const int e0s = (threadIdx.x & 7) * 8; // 0,8,...,56

  // ---- load Q A-fragments (two E-chunks of 16x32 f16), scaled by QSCALE ----
  v16h aQ[2];
  {
    const float* qrow = Q + (((size_t)b * L + (qbase + lane16)) * H + h) * E;
#pragma unroll
    for (int c = 0; c < 2; ++c) {
      v16h a;
#pragma unroll
      for (int i = 0; i < 8; ++i)
        a[i] = (_Float16)(qrow[c * 32 + off1 + i] * QSCALE);
#pragma unroll
      for (int i = 0; i < 8; ++i)
        a[8 + i] = (_Float16)(qrow[c * 32 + off2 + i] * QSCALE);
      aQ[c] = a;
    }
  }

  // ---- per-query-row Gaussian params (base-2) + online-softmax state ----
  const float* sgbh = sg_arr + (size_t)bh * L;
  float di[8], n2s2[8], lcn2[8], mrun[8], lrun[8];
#pragma unroll
  for (int r = 0; r < 8; ++r) {
    int irow = qbase + r + 8 * hi;     // C-layout: VGPR r <-> row r+8*hi
    float s = sgbh[irow];
    di[r]   = (float)irow + s;
    n2s2[r] = -LOG2E / (2.0f * s * s);            // exp2 coefficient
    lcn2[r] = flog2(INV_SQRT_2PI) - flog2(s);     // log2(1/(sqrt(2pi)*s))
    mrun[r] = NEG_BIG;
    lrun[r] = 0.0f;
  }
  v8f oacc[4] = {};                    // 16x64 fp32 output accumulator

  const int ntiles = tileT * 2 + 2;    // causal: keys up to tile end
  for (int t = 0; t < ntiles; ++t) {
    const int j0 = t * 32;
    __syncthreads();
    // ---- cooperative stage: K row-major, V transposed (packed h2 stores) ---
    {
      const float* k0 = Km + (((size_t)b * L + (j0 + 2 * rp)) * H + h) * E + e0s;
      const float* k1 = k0 + (size_t)H * E;
      const float* v0 = Vm + (((size_t)b * L + (j0 + 2 * rp)) * H + h) * E + e0s;
      const float* v1 = v0 + (size_t)H * E;
      float kf0[8], kf1[8], vf0[8], vf1[8];
      *(float4*)&kf0[0] = ((const float4*)k0)[0];
      *(float4*)&kf0[4] = ((const float4*)k0)[1];
      *(float4*)&kf1[0] = ((const float4*)k1)[0];
      *(float4*)&kf1[4] = ((const float4*)k1)[1];
      *(float4*)&vf0[0] = ((const float4*)v0)[0];
      *(float4*)&vf0[4] = ((const float4*)v0)[1];
      *(float4*)&vf1[0] = ((const float4*)v1)[0];
      *(float4*)&vf1[4] = ((const float4*)v1)[1];
      h8 hk0, hk1;
#pragma unroll
      for (int i = 0; i < 8; ++i) hk0[i] = (_Float16)kf0[i];
#pragma unroll
      for (int i = 0; i < 8; ++i) hk1[i] = (_Float16)kf1[i];
      *(h8*)&Kt[(2 * rp) * KT_STRIDE + e0s] = hk0;
      *(h8*)&Kt[(2 * rp + 1) * KT_STRIDE + e0s] = hk1;
#pragma unroll
      for (int i = 0; i < 8; ++i) {
        h2 p;
        p[0] = (_Float16)vf0[i];
        p[1] = (_Float16)vf1[i];
        *(h2*)&VtT[(e0s + i) * VT_STRIDE + 2 * rp] = p;
      }
      if (t + 1 < ntiles) {            // global_prefetch_b8 next tile
        __builtin_prefetch(k0 + (size_t)32 * H * E, 0, 1);
        __builtin_prefetch(v0 + (size_t)32 * H * E, 0, 1);
      }
    }
    __syncthreads();
    if (j0 > qbase + 15) continue;     // wave-uniform; rejoins loop-top barrier

    // ---- S' = (QSCALE*Q) K^T : base-2 logits, two 16x16 fp32 fragments ----
    v8f svec[2];
#pragma unroll
    for (int nh = 0; nh < 2; ++nh) {
      v8f acc = {};
#pragma unroll
      for (int c = 0; c < 2; ++c) {    // E-chunks (K dim of WMMA)
        const _Float16* kp =
            &Kt[(nh * 16 + lane16) * KT_STRIDE + c * 32 + (hi ? 16 : 0)];
        h8 blo = *(const h8*)kp;
        h8 bhi = *(const h8*)(kp + 8);
        v16h bfrag = __builtin_shufflevector(
            blo, bhi, 0, 1, 2, 3, 4, 5, 6, 7, 8, 9, 10, 11, 12, 13, 14, 15);
        acc = __builtin_amdgcn_wmma_f32_16x16x32_f16(false, aQ[c], false, bfrag,
                                                     (short)0, acc, false, false);
      }
      svec[nh] = acc;
    }

    // ---- Gaussian weight (exp2), causal mask, base-2 online softmax ----
    int jg0 = j0 + lane16;
    int jg1 = j0 + 16 + lane16;
    float dj0 = (float)jg0 + sgbh[jg0];
    float dj1 = (float)jg1 + sgbh[jg1];

    float p0[8], p1[8];
#pragma unroll
    for (int r = 0; r < 8; ++r) {
      int irow = qbase + r + 8 * hi;
      float t0 = di[r] - dj0;
      float w0 = fexp2(fmaf(t0 * t0, n2s2[r], lcn2[r]));
      float l0 = (jg0 <= irow) ? (svec[0][r] * w0) : NEG_BIG;
      float t1 = di[r] - dj1;
      float w1 = fexp2(fmaf(t1 * t1, n2s2[r], lcn2[r]));
      float l1 = (jg1 <= irow) ? (svec[1][r] * w1) : NEG_BIG;

      float mx = rowmax16(fmaxf(l0, l1));
      float mnew = fmaxf(mrun[r], mx);
      float alpha = fexp2(mrun[r] - mnew);
      float e0 = fexp2(l0 - mnew);
      float e1 = fexp2(l1 - mnew);
      float sum = rowsum16(e0 + e1);
      lrun[r] = lrun[r] * alpha + sum;
      mrun[r] = mnew;
      p0[r] = e0;
      p1[r] = e1;
#pragma unroll
      for (int c = 0; c < 4; ++c) oacc[c][r] *= alpha;
    }

    // ---- P (C-layout fp32) -> LDS -> A-layout f16 fragment ----
    _Float16* pw = &Pw[wave][0];
#pragma unroll
    for (int r = 0; r < 8; ++r) {
      pw[(r + 8 * hi) * PW_STRIDE + lane16]      = (_Float16)p0[r];
      pw[(r + 8 * hi) * PW_STRIDE + 16 + lane16] = (_Float16)p1[r];
    }
    asm volatile("s_wait_dscnt 0" ::: "memory");  // LDS RAW fence (same wave)
    const _Float16* prow = pw + lane16 * PW_STRIDE;
    h8 plo = *(const h8*)(prow + off1);
    h8 phi = *(const h8*)(prow + off2);
    v16h pa = __builtin_shufflevector(
        plo, phi, 0, 1, 2, 3, 4, 5, 6, 7, 8, 9, 10, 11, 12, 13, 14, 15);

    // ---- O += P * V  (four N-chunks of E) ----
#pragma unroll
    for (int c = 0; c < 4; ++c) {
      const _Float16* vp = &VtT[(c * 16 + lane16) * VT_STRIDE + (hi ? 16 : 0)];
      h8 vlo = *(const h8*)vp;
      h8 vhi = *(const h8*)(vp + 8);
      v16h vb = __builtin_shufflevector(
          vlo, vhi, 0, 1, 2, 3, 4, 5, 6, 7, 8, 9, 10, 11, 12, 13, 14, 15);
      oacc[c] = __builtin_amdgcn_wmma_f32_16x16x32_f16(false, pa, false, vb,
                                                       (short)0, oacc[c],
                                                       false, false);
    }
  }

  // ---- normalize and write out[b, i, h, e] ----
#pragma unroll
  for (int r = 0; r < 8; ++r) {
    int irow = qbase + r + 8 * hi;
    float inv = 1.0f / lrun[r];
    float* orow = Out + (((size_t)b * L + irow) * H + h) * E;
#pragma unroll
    for (int c = 0; c < 4; ++c) orow[c * 16 + lane16] = oacc[c][r] * inv;
  }
}

// ---------------------------------------------------------------------------
extern "C" void kernel_launch(void* const* d_in, const int* in_sizes, int n_in,
                              void* d_out, int out_size, void* d_ws,
                              size_t ws_size, hipStream_t stream) {
  (void)in_sizes; (void)n_in; (void)out_size; (void)ws_size;
  const int B = 2, L = 2048, H = 8;

  const float* Q     = (const float*)d_in[0];
  const float* K     = (const float*)d_in[1];
  const float* V     = (const float*)d_in[2];
  const float* sigma = (const float*)d_in[3];
  float* out = (float*)d_out;
  float* sg  = (float*)d_ws;  // B*H*L floats = 128 KB scratch

  int total = B * H * L;
  sg_precompute_kernel<<<(total + 255) / 256, 256, 0, stream>>>(sigma, sg, B, L,
                                                                H);
  dim3 grid(L / 64, B * H);
  anomaly_attn_kernel<<<grid, 128, 0, stream>>>(Q, K, V, sg, out, B, L, H);
}
